// MAHALANOBIS_LOSS_41936060678350
// MI455X (gfx1250) — matrix operand order repaired and checked by reading, not compiled, not run on hardware
//
#include <hip/hip_runtime.h>
#include <math.h>

// Problem constants (match reference)
#define B_ROWS   8192
#define D_DIM    2048

// Tiling
#define BM       128
#define BN       128
#define BK       32
#define SA       40            // LDS row stride in ushorts (BK + 8 pad) -> 80B, 16B aligned
#define SB       40
#define NTHREADS 256           // 8 waves (wave32)

typedef __attribute__((ext_vector_type(16))) __bf16 v16bf;
typedef __attribute__((ext_vector_type(8)))  float  v8f;

union Frag {
    uint4 q[2];   // 2 x 16B LDS loads
    v16bf v;      // 16 bf16 -> 8 VGPRs
};

__device__ __forceinline__ unsigned short f32_to_bf16_rne(float f) {
    unsigned int u = __float_as_uint(f);
    u += 0x7FFFu + ((u >> 16) & 1u);           // round-to-nearest-even
    return (unsigned short)(u >> 16);
}
__device__ __forceinline__ float bf16_to_f32(unsigned short h) {
    return __uint_as_float(((unsigned int)h) << 16);
}
__device__ __forceinline__ unsigned int pack2(unsigned short a, unsigned short b) {
    return (unsigned int)a | ((unsigned int)b << 16);
}

// Split x into bf16 hi + bf16 lo (x ~= hi + lo, ~16 mantissa bits captured)
__device__ __forceinline__ void split_bf16(float x, unsigned short& h, unsigned short& l) {
    h = f32_to_bf16_rne(x);
    l = f32_to_bf16_rne(x - bf16_to_f32(h));
}

__global__ void __launch_bounds__(256)
mahal_zero_q(float* __restrict__ q) {
    q[blockIdx.x * 256 + threadIdx.x] = 0.0f;
}

// Fused: q[b] += sum_n ( sum_k delta[b,k] * S[n,k] ) * delta[b,n]   (S symmetric)
__global__ void __launch_bounds__(NTHREADS, 1)
mahal_gemm_kernel(const float* __restrict__ orig,
                  const float* __restrict__ recon,
                  const float* __restrict__ sinv,
                  float* __restrict__ q) {
    __shared__ alignas(16) unsigned short As_hi[BM * SA];
    __shared__ alignas(16) unsigned short As_lo[BM * SA];
    __shared__ alignas(16) unsigned short Bs_hi[BN * SB];
    __shared__ alignas(16) unsigned short Bs_lo[BN * SB];

    const int tid  = threadIdx.x;
    const int bn0  = blockIdx.x * BN;       // N tile base (output column / second delta index)
    const int bm0  = blockIdx.y * BM;       // M tile base (batch row)

    const int lane = tid & 31;
    const int wave = tid >> 5;
    const int wm   = wave & 3;              // 4 waves along M  -> 32 rows each
    const int wn   = wave >> 2;             // 2 waves along N  -> 64 cols each
    const int hi   = lane >> 4;             // half-wave select
    const int lr   = lane & 15;

    // cooperative loader mapping: 32 rows x 8 float4 per pass, 4 passes -> 128 rows
    const int lrow  = tid >> 3;             // 0..31
    const int lquad = tid & 7;              // 0..7 -> k offset lquad*4

    v8f c[2][4];
    {
        v8f z = {};
        #pragma unroll
        for (int i = 0; i < 2; ++i)
            #pragma unroll
            for (int j = 0; j < 4; ++j)
                c[i][j] = z;
    }

    for (int k0 = 0; k0 < D_DIM; k0 += BK) {
        // Prefetch next K tile into cache hierarchy (global_prefetch_b8)
        if (k0 + BK < D_DIM) {
            __builtin_prefetch(orig  + (size_t)(bm0 + lrow) * D_DIM + (k0 + BK) + lquad * 4, 0, 3);
            __builtin_prefetch(recon + (size_t)(bm0 + lrow) * D_DIM + (k0 + BK) + lquad * 4, 0, 3);
            __builtin_prefetch(sinv  + (size_t)(bn0 + lrow) * D_DIM + (k0 + BK) + lquad * 4, 0, 3);
        }

        // ---- fill LDS: A = split-bf16(delta[M, k0:k0+32]); B = split-bf16(S[N, k0:k0+32]) ----
        #pragma unroll
        for (int r = 0; r < 4; ++r) {
            const int row = lrow + r * 32;

            const float4 o4 = *(const float4*)(orig  + (size_t)(bm0 + row) * D_DIM + k0 + lquad * 4);
            const float4 r4 = *(const float4*)(recon + (size_t)(bm0 + row) * D_DIM + k0 + lquad * 4);
            float d0 = o4.x - r4.x, d1 = o4.y - r4.y, d2 = o4.z - r4.z, d3 = o4.w - r4.w;
            unsigned short h0, h1, h2, h3, l0, l1, l2, l3;
            split_bf16(d0, h0, l0); split_bf16(d1, h1, l1);
            split_bf16(d2, h2, l2); split_bf16(d3, h3, l3);
            uint2 ph; ph.x = pack2(h0, h1); ph.y = pack2(h2, h3);
            uint2 pl; pl.x = pack2(l0, l1); pl.y = pack2(l2, l3);
            *(uint2*)&As_hi[row * SA + lquad * 4] = ph;
            *(uint2*)&As_lo[row * SA + lquad * 4] = pl;

            // S_inv is symmetric: read row n contiguously along k
            const float4 s4 = *(const float4*)(sinv + (size_t)(bn0 + row) * D_DIM + k0 + lquad * 4);
            split_bf16(s4.x, h0, l0); split_bf16(s4.y, h1, l1);
            split_bf16(s4.z, h2, l2); split_bf16(s4.w, h3, l3);
            ph.x = pack2(h0, h1); ph.y = pack2(h2, h3);
            pl.x = pack2(l0, l1); pl.y = pack2(l2, l3);
            *(uint2*)&Bs_hi[row * SB + lquad * 4] = ph;
            *(uint2*)&Bs_lo[row * SB + lquad * 4] = pl;
        }
        __syncthreads();

        // ---- load fragments (contiguous k order, same for A and B) ----
        // A (16x32, 16-bit): lane(row M=lr), elem e -> K = 16*hi + e (16 consecutive k)
        Frag ahi[2], alo[2];
        #pragma unroll
        for (int i = 0; i < 2; ++i) {
            const int arow = wm * 32 + i * 16 + lr;
            ahi[i].q[0] = *(const uint4*)&As_hi[arow * SA + hi * 16];
            ahi[i].q[1] = *(const uint4*)&As_hi[arow * SA + hi * 16 + 8];
            alo[i].q[0] = *(const uint4*)&As_lo[arow * SA + hi * 16];
            alo[i].q[1] = *(const uint4*)&As_lo[arow * SA + hi * 16 + 8];
        }
        // B (32x16, 16-bit): lane(col N=lr), elem e -> K = 16*hi + e (16 consecutive k)
        Frag bhi[4], blo[4];
        #pragma unroll
        for (int j = 0; j < 4; ++j) {
            const int brow = wn * 64 + j * 16 + lr;
            bhi[j].q[0] = *(const uint4*)&Bs_hi[brow * SB + hi * 16];
            bhi[j].q[1] = *(const uint4*)&Bs_hi[brow * SB + hi * 16 + 8];
            blo[j].q[0] = *(const uint4*)&Bs_lo[brow * SB + hi * 16];
            blo[j].q[1] = *(const uint4*)&Bs_lo[brow * SB + hi * 16 + 8];
        }

        // ---- WMMA: c += Ahi*Bhi + Alo*Bhi + Ahi*Blo  (drop lo*lo) ----
        #pragma unroll
        for (int i = 0; i < 2; ++i) {
            #pragma unroll
            for (int j = 0; j < 4; ++j) {
                c[i][j] = __builtin_amdgcn_wmma_f32_16x16x32_bf16(
                    false, ahi[i].v, false, bhi[j].v, (short)0, c[i][j], false, false);
                c[i][j] = __builtin_amdgcn_wmma_f32_16x16x32_bf16(
                    false, alo[i].v, false, bhi[j].v, (short)0, c[i][j], false, false);
                c[i][j] = __builtin_amdgcn_wmma_f32_16x16x32_bf16(
                    false, ahi[i].v, false, blo[j].v, (short)0, c[i][j], false, false);
            }
        }
        __syncthreads();
    }

    // ---- epilogue: q[m] += sum_n  C[m,n] * delta[m,n]  (f32 delta recomputed, L2 hits) ----
    // C layout: VGPR v -> M = v + 8*hi ; N = lr (per 16x16 tile)
    #pragma unroll
    for (int i = 0; i < 2; ++i) {
        #pragma unroll
        for (int v = 0; v < 8; ++v) {
            const int m = bm0 + wm * 32 + i * 16 + hi * 8 + v;
            const float* orow = orig  + (size_t)m * D_DIM + bn0 + wn * 64 + lr;
            const float* rrow = recon + (size_t)m * D_DIM + bn0 + wn * 64 + lr;
            float part = 0.0f;
            #pragma unroll
            for (int j = 0; j < 4; ++j) {
                const float dl = orow[j * 16] - rrow[j * 16];
                part += c[i][j][v] * dl;
            }
            // reduce across the 16 lanes (same hi-half) that share row m
            #pragma unroll
            for (int mask = 1; mask < 16; mask <<= 1)
                part += __shfl_xor(part, mask, 32);
            if (lr == 0) atomicAdd(&q[m], part);
        }
    }
}

__global__ void __launch_bounds__(256)
mahal_finalize(const float* __restrict__ q, float* __restrict__ out) {
    __shared__ float red[256];
    float s = 0.0f;
    for (int i = threadIdx.x; i < B_ROWS; i += 256)
        s += sqrtf(q[i]);
    red[threadIdx.x] = s;
    __syncthreads();
    #pragma unroll
    for (int w = 128; w > 0; w >>= 1) {
        if (threadIdx.x < w) red[threadIdx.x] += red[threadIdx.x + w];
        __syncthreads();
    }
    if (threadIdx.x == 0) out[0] = red[0] / (float)B_ROWS;
}

extern "C" void kernel_launch(void* const* d_in, const int* in_sizes, int n_in,
                              void* d_out, int out_size, void* d_ws, size_t ws_size,
                              hipStream_t stream) {
    (void)in_sizes; (void)n_in; (void)out_size; (void)ws_size;
    const float* orig  = (const float*)d_in[0];
    const float* recon = (const float*)d_in[1];
    const float* sinv  = (const float*)d_in[2];
    float* qbuf = (float*)d_ws;                 // B_ROWS floats = 32 KB scratch
    float* out  = (float*)d_out;

    hipLaunchKernelGGL(mahal_zero_q, dim3(B_ROWS / 256), dim3(256), 0, stream, qbuf);

    dim3 grid(D_DIM / BN, B_ROWS / BM);         // 16 x 64 blocks
    hipLaunchKernelGGL(mahal_gemm_kernel, grid, dim3(NTHREADS), 0, stream,
                       orig, recon, sinv, qbuf);

    hipLaunchKernelGGL(mahal_finalize, dim3(1), dim3(256), 0, stream, qbuf, out);
}
